// GNNBaseline_69157563400861
// MI455X (gfx1250) — matrix operand order repaired and checked by reading, not compiled, not run on hardware
//
#include <hip/hip_runtime.h>

#define N_NODES 50000
#define N_EDGES 800000

typedef __attribute__((ext_vector_type(16))) __bf16 v16bf;
typedef __attribute__((ext_vector_type(8)))  float  v8f;

// ---------- monotonic float<->uint encoding for atomic segment-max ----------
__device__ __forceinline__ unsigned fenc(float f) {
  unsigned u = __float_as_uint(f);
  return (u & 0x80000000u) ? ~u : (u | 0x80000000u);
}
__device__ __forceinline__ float fdec(unsigned u) {
  return (u & 0x80000000u) ? __uint_as_float(u & 0x7FFFFFFFu)
                           : __uint_as_float(~u);
}

__device__ __forceinline__ void pack4(v16bf& v, int base, float4 p) {
  v[base + 0] = (__bf16)p.x; v[base + 1] = (__bf16)p.y;
  v[base + 2] = (__bf16)p.z; v[base + 3] = (__bf16)p.w;
}

// ---------- WMMA GEMM: Y[N,KOUT] = X[N,64] @ W[64,KOUT] (+bias) ----------
// One wave owns a 16-row stripe and all KOUT/16 column tiles, so the two A
// fragments (K=64 as two bf16 k-steps of 32) are loaded once and reused.
// Fragment layouts per ISA 7.12.2:
//   A 16x32 bf16 : lanes 0-15 rows M=0..15, K={kb..kb+7, kb+16..kb+23},
//                  lanes 16-31 same rows, kb+=8  -> two aligned float4 pairs.
//   B 32x16 bf16 : lane l holds column N=l&15, K as for A  -> 16 scalar loads.
//   C/D 16x16 f32: lane<16 -> M=v, lane>=16 -> M=v+8; N = lane&15.
template <int KOUT>
__global__ void gemm_wmma(const float* __restrict__ X,
                          const float* __restrict__ W,
                          const float* __restrict__ bias,
                          float* __restrict__ Y) {
  constexpr int CT = KOUT / 16;
  int wave = (blockIdx.x * blockDim.x + threadIdx.x) >> 5;
  int lane = threadIdx.x & 31;
  if (wave >= (N_NODES >> 4)) return;
  int m0   = wave << 4;
  int half = lane >> 4;
  int l16  = lane & 15;

  // A fragments (both k-steps), vectorized b128 loads
  const float4* xrow4 = (const float4*)(X + (size_t)(m0 + l16) * 64);
  v16bf afrag[2];
#pragma unroll
  for (int ks = 0; ks < 2; ++ks) {
    int kb = ks * 32 + half * 8;              // multiple of 8 -> float4 aligned
    pack4(afrag[ks], 0,  xrow4[(kb >> 2)]);
    pack4(afrag[ks], 4,  xrow4[(kb >> 2) + 1]);
    pack4(afrag[ks], 8,  xrow4[((kb + 16) >> 2)]);
    pack4(afrag[ks], 12, xrow4[((kb + 16) >> 2) + 1]);
  }

  const float* wrow0 = W + (size_t)(half * 8) * KOUT + l16;        // ks=0: K=8h+{0..7,16..23}, N=l16
  const float* wrow1 = W + (size_t)(32 + half * 8) * KOUT + l16;   // ks=1: K=32+8h+{0..7,16..23}

  v8f acc[CT] = {};
#pragma unroll
  for (int ct = 0; ct < CT; ++ct) {
    int n4 = ct << 2;  // n0/4
#pragma unroll
    for (int ks = 0; ks < 2; ++ks) {
      const float* wr = ks ? wrow1 : wrow0;
      v16bf b;
#pragma unroll
      for (int s = 0; s < 16; ++s)
        b[s] = (__bf16)wr[(size_t)(s < 8 ? s : s + 8) * KOUT + n4 * 4];
      acc[ct] = __builtin_amdgcn_wmma_f32_16x16x32_bf16(
          /*neg_a=*/false, afrag[ks], /*neg_b=*/false, b,
          /*c_mod=*/(short)0, acc[ct], /*reuse_a=*/false, /*reuse_b=*/false);
    }
  }

#pragma unroll
  for (int ct = 0; ct < CT; ++ct) {
    float badd = bias ? bias[ct * 16 + l16] : 0.0f;
#pragma unroll
    for (int v = 0; v < 8; ++v) {
      int row = m0 + v + half * 8;
      Y[(size_t)row * KOUT + ct * 16 + l16] = acc[ct][v] + badd;
    }
  }
}

// ---------- per-layer init: zero accumulator & denom, -inf running max ----------
__global__ void init_layer(float* ACC, float* DEN, unsigned* M, int accN, int nh) {
  int i = blockIdx.x * blockDim.x + threadIdx.x;
  if (i < accN) ACC[i] = 0.0f;
  if (i < nh) { DEN[i] = 0.0f; M[i] = 0x007FFFFFu; /* fenc(-inf) */ }
}

// ---------- alpha_src / alpha_dst = <xw[n,h,:], a_src/dst[h,:]> ----------
__global__ void alpha_node(const float* __restrict__ XW,
                           const float* __restrict__ asrc,
                           const float* __restrict__ adst,
                           float* __restrict__ AS, float* __restrict__ AD,
                           int H) {
  int idx = blockIdx.x * blockDim.x + threadIdx.x;
  if (idx >= N_NODES * H) return;
  int n = idx / H, hh = idx % H;
  const float4* xp = (const float4*)(XW + (size_t)n * H * 64 + hh * 64);
  const float4* s  = (const float4*)(asrc + hh * 64);
  const float4* d  = (const float4*)(adst + hh * 64);
  float ss = 0.0f, dd = 0.0f;
#pragma unroll 4
  for (int q = 0; q < 16; ++q) {
    float4 v = xp[q], a = s[q], b = d[q];
    ss += v.x * a.x + v.y * a.y + v.z * a.z + v.w * a.w;
    dd += v.x * b.x + v.y * b.y + v.z * b.z + v.w * b.w;
  }
  AS[idx] = ss; AD[idx] = dd;
}

// ---------- folded edge attention vector: VE[k,h] = sum_c We[k,h,c]*ae[h,c] ----------
__global__ void ve_kernel(const float* __restrict__ We, const float* __restrict__ ae,
                          float* __restrict__ VE, int H) {
  int idx = threadIdx.x;
  if (idx >= 16 * H) return;
  int k = idx / H, hh = idx % H;
  float s = 0.0f;
  for (int cc = 0; cc < 64; ++cc)
    s += We[(size_t)k * H * 64 + hh * 64 + cc] * ae[hh * 64 + cc];
  VE[k * H + hh] = s;
}

// ---------- edge pass 1: logits, leaky-relu, atomic segment max ----------
__global__ void edge_p1(const float* __restrict__ wfeat,
                        const int* __restrict__ src, const int* __restrict__ dst,
                        const float* __restrict__ VE,
                        const float* __restrict__ AS, const float* __restrict__ AD,
                        float* __restrict__ EAL, unsigned* __restrict__ M, int H) {
  int e = blockIdx.x * blockDim.x + threadIdx.x;
  if (e >= N_EDGES) return;
  int s = src[e], d = dst[e];
  float wl[16];
  const float4* wp = (const float4*)(wfeat + (size_t)e * 16);
#pragma unroll
  for (int q = 0; q < 4; ++q) {
    float4 p = wp[q];
    wl[4 * q + 0] = p.x; wl[4 * q + 1] = p.y;
    wl[4 * q + 2] = p.z; wl[4 * q + 3] = p.w;
  }
  for (int hh = 0; hh < H; ++hh) {
    float ae = 0.0f;
#pragma unroll
    for (int k = 0; k < 16; ++k) ae += wl[k] * VE[k * H + hh];
    float a = AS[s * H + hh] + AD[d * H + hh] + ae;
    a = (a >= 0.0f) ? a : 0.2f * a;           // leaky_relu(0.2)
    EAL[(size_t)e * H + hh] = a;
    atomicMax(&M[d * H + hh], fenc(a));
  }
}

// ---------- edge pass 2: exp(alpha - max), atomic denom sum ----------
__global__ void edge_p2(float* __restrict__ EAL,
                        const int* __restrict__ dst,
                        const unsigned* __restrict__ M,
                        float* __restrict__ DEN, int H) {
  int e = blockIdx.x * blockDim.x + threadIdx.x;
  if (e >= N_EDGES) return;
  int d = dst[e];
  for (int hh = 0; hh < H; ++hh) {
    float a  = EAL[(size_t)e * H + hh];
    float ex = __expf(a - fdec(M[d * H + hh]));
    EAL[(size_t)e * H + hh] = ex;
    atomicAdd(&DEN[d * H + hh], ex);
  }
}

// ---------- edge pass 3: message scatter (wave per edge, L2-resident atomics) ----------
__global__ void edge_p3(const float* __restrict__ XW,
                        const float* __restrict__ EAL,
                        const float* __restrict__ DEN,
                        const int* __restrict__ src, const int* __restrict__ dst,
                        float* __restrict__ ACC, int H) {
  int wave = (blockIdx.x * blockDim.x + threadIdx.x) >> 5;
  int lane = threadIdx.x & 31;
  if (wave >= N_EDGES) return;
  int s = src[wave], d = dst[wave];
  for (int hh = 0; hh < H; ++hh) {
    float coef = EAL[(size_t)wave * H + hh] / (DEN[d * H + hh] + 1e-16f);
    size_t sb = (size_t)s * H * 64 + hh * 64;
    size_t db = (size_t)d * H * 64 + hh * 64;
    atomicAdd(&ACC[db + lane],      coef * XW[sb + lane]);
    atomicAdd(&ACC[db + lane + 32], coef * XW[sb + lane + 32]);
  }
}

// ---------- head-mean + bias + SiLU ----------
__global__ void finalize_k(const float* __restrict__ ACC, const float* __restrict__ bvec,
                           float* __restrict__ Xout, int H) {
  int idx = blockIdx.x * blockDim.x + threadIdx.x;
  if (idx >= N_NODES * 64) return;
  int n = idx >> 6, c = idx & 63;
  float v = (H == 2) ? 0.5f * (ACC[(size_t)n * 128 + c] + ACC[(size_t)n * 128 + 64 + c])
                     : ACC[(size_t)n * 64 + c];
  v += bvec[c];
  Xout[idx] = v * (1.0f / (1.0f + __expf(-v)));   // silu
}

__global__ void copy_f32_v4(const float4* __restrict__ in, float4* __restrict__ out,
                            int n4) {
  int i = blockIdx.x * blockDim.x + threadIdx.x;
  if (i < n4) out[i] = in[i];
}

// ---------- one GAT layer ----------
static void run_layer(const float* xin, const float* W, const float* asrc,
                      const float* adst, const float* We, const float* ae,
                      const float* bvec, int H,
                      const int* src, const int* dst, const float* wfeat,
                      float* XW, float* ACC, float* AS, float* AD,
                      unsigned* M, float* DEN, float* EAL, float* VE,
                      float* xout, hipStream_t stream) {
  int HC   = H * 64;
  int accN = N_NODES * HC;
  int nh   = N_NODES * H;
  init_layer<<<(accN + 255) / 256, 256, 0, stream>>>(ACC, DEN, M, accN, nh);
  int gblocks = ((N_NODES / 16) + 3) / 4;     // 4 waves/block, wave per 16-row stripe
  if (H == 2)
    gemm_wmma<128><<<gblocks, 128, 0, stream>>>(xin, W, nullptr, XW);
  else
    gemm_wmma<64><<<gblocks, 128, 0, stream>>>(xin, W, nullptr, XW);
  alpha_node<<<(nh + 255) / 256, 256, 0, stream>>>(XW, asrc, adst, AS, AD, H);
  ve_kernel<<<1, 32, 0, stream>>>(We, ae, VE, H);
  edge_p1<<<(N_EDGES + 255) / 256, 256, 0, stream>>>(wfeat, src, dst, VE, AS, AD, EAL, M, H);
  edge_p2<<<(N_EDGES + 255) / 256, 256, 0, stream>>>(EAL, dst, M, DEN, H);
  edge_p3<<<(N_EDGES * 32) / 256, 256, 0, stream>>>(XW, EAL, DEN, src, dst, ACC, H);
  finalize_k<<<(N_NODES * 64) / 256, 256, 0, stream>>>(ACC, bvec, xout, H);
}

extern "C" void kernel_launch(void* const* d_in, const int* in_sizes, int n_in,
                              void* d_out, int out_size, void* d_ws, size_t ws_size,
                              hipStream_t stream) {
  (void)in_sizes; (void)n_in; (void)out_size; (void)ws_size;
  const float* h   = (const float*)d_in[0];
  const float* e   = (const float*)d_in[1];
  const int*   ei  = (const int*)  d_in[2];
  const float* w   = (const float*)d_in[3];
  const float* W0  = (const float*)d_in[4];
  const float* as0 = (const float*)d_in[5];
  const float* ad0 = (const float*)d_in[6];
  const float* We0 = (const float*)d_in[7];
  const float* ae0 = (const float*)d_in[8];
  const float* b0  = (const float*)d_in[9];
  const float* W1  = (const float*)d_in[10];
  const float* as1 = (const float*)d_in[11];
  const float* ad1 = (const float*)d_in[12];
  const float* We1 = (const float*)d_in[13];
  const float* ae1 = (const float*)d_in[14];
  const float* b1  = (const float*)d_in[15];
  const float* W2  = (const float*)d_in[16];
  const float* as2 = (const float*)d_in[17];
  const float* ad2 = (const float*)d_in[18];
  const float* We2 = (const float*)d_in[19];
  const float* ae2 = (const float*)d_in[20];
  const float* b2  = (const float*)d_in[21];
  const float* Wr  = (const float*)d_in[22];
  const float* br  = (const float*)d_in[23];
  float* out = (float*)d_out;

  const int* src = ei;
  const int* dst = ei + N_EDGES;

  // workspace layout (floats)
  float*    ws  = (float*)d_ws;
  float*    XW  = ws;                        // N*128
  float*    ACC = ws + 6400000;              // N*128
  float*    X1  = ws + 12800000;             // N*64
  float*    X2  = ws + 16000000;             // N*64
  float*    AS  = ws + 19200000;             // N*2
  float*    AD  = ws + 19300000;             // N*2
  unsigned* M   = (unsigned*)(ws + 19400000);// N*2
  float*    DEN = ws + 19500000;             // N*2
  float*    EAL = ws + 19600000;             // E*2
  float*    VE  = ws + 21200000;             // 32

  // second tuple output: e passes through unchanged (b128 copy)
  copy_f32_v4<<<(N_EDGES * 16 / 4) / 256, 256, 0, stream>>>(
      (const float4*)e, (float4*)(out + (size_t)N_NODES * 64), N_EDGES * 16 / 4);

  run_layer(h,  W0, as0, ad0, We0, ae0, b0, 2, src, dst, w,
            XW, ACC, AS, AD, M, DEN, EAL, VE, X1, stream);
  run_layer(X1, W1, as1, ad1, We1, ae1, b1, 1, src, dst, w,
            XW, ACC, AS, AD, M, DEN, EAL, VE, X2, stream);
  run_layer(X2, W2, as2, ad2, We2, ae2, b2, 1, src, dst, w,
            XW, ACC, AS, AD, M, DEN, EAL, VE, X1, stream);

  // readout: out[N,64] = X1 @ Wr + br  (bias fused into WMMA epilogue)
  int gblocks = ((N_NODES / 16) + 3) / 4;
  gemm_wmma<64><<<gblocks, 128, 0, stream>>>(X1, Wr, br, out);
}
